// S2NeuralFlows_9431748182123
// MI455X (gfx1250) — matrix operand order repaired and checked by reading, not compiled, not run on hardware
//
#include <hip/hip_runtime.h>
#include <hip/hip_bf16.h>
#include <math.h>

typedef __attribute__((ext_vector_type(16))) _Float16 v16h;
typedef __attribute__((ext_vector_type(8)))  _Float16 v8h;
typedef __attribute__((ext_vector_type(8)))  float    v8f;

#define KBINS 50
#define HID   256
#define NLAY  5
#define PI_F  3.14159265358979323846f

// ---------------------------------------------------------------------------
// Prep: W2 f32 [l][j=256][k=50]  ->  W2t f16 [l][n=64][j=256]  (n>=50 zeroed)
// Makes the WMMA B-fragment a contiguous 32B read per lane (K along memory).
// ---------------------------------------------------------------------------
__global__ void w2_transpose_kernel(const float* __restrict__ W2,
                                    _Float16* __restrict__ W2t) {
    int idx = blockIdx.x * blockDim.x + threadIdx.x;  // < 5*64*256 = 81920
    if (idx >= NLAY * 64 * HID) return;
    int l = idx >> 14;          // / (64*256)
    int r = idx & 16383;
    int n = r >> 8;             // 0..63
    int j = r & 255;            // 0..255
    float v = (n < KBINS) ? W2[l * (HID * KBINS) + j * KBINS + n] : 0.0f;
    W2t[idx] = (_Float16)v;
}

// ---------------------------------------------------------------------------
// Main flow kernel: 128 threads = 4 waves, each wave owns 32 points.
// Per layer: build A = relu(z*W1+b1) in registers, WMMA against W2t tiles,
// transpose D through wave-private LDS, then softmax + linear spline.
// ---------------------------------------------------------------------------
__global__ __launch_bounds__(128) void flow_kernel(
    const float* __restrict__ x,
    const float* __restrict__ W1,
    const float* __restrict__ b1,
    const _Float16* __restrict__ W2t,
    const float* __restrict__ b2,
    float* __restrict__ out)
{
    __shared__ float sW1[HID];
    __shared__ float sB1[HID];
    __shared__ float sB2[KBINS];
    __shared__ float sLog[4][32 * 65];   // per-wave logits tile, pad 65 vs banks

    const int tid    = threadIdx.x;
    const int wid    = tid >> 5;
    const int lane   = tid & 31;
    const int lane16 = lane & 15;
    const int hi     = lane >> 4;        // 0: lanes 0-15, 1: lanes 16-31
    const int p      = blockIdx.x * 128 + tid;

    // ---- circle2box ----
    float x0 = x[2 * p + 0];
    float x1 = x[2 * p + 1];
    float xn0 = x0 * (1.0f / PI_F);
    float xn1 = x1 * (1.0f / PI_F);
    float n2  = xn0 * xn0 + xn1 * xn1;
    bool  inside = n2 < 1.0f;
    float den = rsqrtf(fmaxf(1.0f - n2, 1e-12f));
    float z0 = inside ? atanf(xn0 * den) * (2.0f / PI_F) : 0.0f;
    float z1 = inside ? atanf(xn1 * den) * (2.0f / PI_F) : 0.0f;

    #pragma unroll 1
    for (int l = 0; l < NLAY; ++l) {
        // ---- stage per-layer constants into LDS ----
        __syncthreads();
        for (int j = tid; j < HID; j += 128) {
            sW1[j] = W1[l * HID + j];
            sB1[j] = b1[l * HID + j];
        }
        if (tid < KBINS) sB2[tid] = b2[l * KBINS + tid];
        __syncthreads();

        const int i_id = 1 - (l & 1);            // ID_DIMS = (1,0,1,0,1)
        float zid = i_id ? z1 : z0;
        float ztr = i_id ? z0 : z1;

        // broadcast z_id of each row of the two 16-row sub-tiles
        float zrow0 = __shfl(zid, lane16, 32);
        float zrow1 = __shfl(zid, 16 + lane16, 32);

        v8f acc0[4] = {};                        // sub-tile 0, 4 N-tiles
        v8f acc1[4] = {};                        // sub-tile 1

        const _Float16* w2l = W2t + l * (64 * HID);
        if (l + 1 < NLAY)                        // warm L2 for next layer's B
            __builtin_prefetch(w2l + 64 * HID + lane * 64, 0, 1);

        #pragma unroll
        for (int kb = 0; kb < 8; ++kb) {         // K = 256 in chunks of 32
            // ---- A fragments (16-bit A 16x32 layout) built in registers ----
            v16h a0, a1;
            #pragma unroll
            for (int i = 0; i < 16; ++i) {
                int rr = i >> 1, cc = i & 1;
                int kbase = (rr < 4) ? (2 * rr) : (16 + 2 * (rr - 4));
                int kk = kb * 32 + i + (hi ? 16 : 0);
                float w  = sW1[kk];
                float bb = sB1[kk];
                a0[i] = (_Float16)fmaxf(zrow0 * w + bb, 0.0f);
                a1[i] = (_Float16)fmaxf(zrow1 * w + bb, 0.0f);
            }
            #pragma unroll
            for (int nb = 0; nb < 4; ++nb) {
                // B fragment: lanes 0-15 -> K=kb*32+0..15, lanes 16-31 -> +16..31
                const v8h* bp = (const v8h*)(w2l + (nb * 16 + lane16) * HID
                                             + kb * 32 + hi * 16);
                v8h blo = bp[0];
                v8h bh2 = bp[1];
                v16h bf = __builtin_shufflevector(blo, bh2,
                           0,1,2,3,4,5,6,7,8,9,10,11,12,13,14,15);
                acc0[nb] = __builtin_amdgcn_wmma_f32_16x16x32_f16(
                               false, a0, false, bf, (short)0, acc0[nb], false, false);
                acc1[nb] = __builtin_amdgcn_wmma_f32_16x16x32_f16(
                               false, a1, false, bf, (short)0, acc1[nb], false, false);
            }
        }

        // ---- transpose D tiles through wave-private LDS ----
        float* lg = &sLog[wid][0];
        #pragma unroll
        for (int nb = 0; nb < 4; ++nb) {
            #pragma unroll
            for (int v = 0; v < 8; ++v) {
                int row = v + hi * 8;
                lg[row * 65 + nb * 16 + lane16]        = acc0[nb][v];
                lg[(16 + row) * 65 + nb * 16 + lane16] = acc1[nb][v];
            }
        }
        __syncthreads();

        // ---- per-point softmax + uniform-bin linear spline ----
        float lv[KBINS];
        float m = -1e30f;
        #pragma unroll
        for (int k = 0; k < KBINS; ++k) {
            lv[k] = lg[lane * 65 + k] + sB2[k];
            m = fmaxf(m, lv[k]);
        }
        float u   = (ztr + 1.0f) * 0.5f;
        float pos = fminf(fmaxf(u, 0.0f), 1.0f - 1e-6f) * (float)KBINS;
        int   idx = (int)pos;
        float alpha = pos - (float)idx;

        float S = 0.0f, pre = 0.0f, ei = 0.0f;
        #pragma unroll
        for (int k = 0; k < KBINS; ++k) {
            float e = __expf(lv[k] - m);
            S += e;
            if (k < idx)  pre += e;
            if (k == idx) ei = e;
        }
        float y = (pre + alpha * ei) / S;        // spline CDF value in [0,1]
        float znew = 2.0f * y - 1.0f;
        if (i_id) z0 = znew; else z1 = znew;     // i_tr == 0 iff i_id == 1
    }

    out[2 * p + 0] = z0;
    out[2 * p + 1] = z1;
}

// ---------------------------------------------------------------------------
extern "C" void kernel_launch(void* const* d_in, const int* in_sizes, int n_in,
                              void* d_out, int out_size, void* d_ws, size_t ws_size,
                              hipStream_t stream) {
    const float* x  = (const float*)d_in[0];
    const float* W1 = (const float*)d_in[1];
    const float* b1 = (const float*)d_in[2];
    const float* W2 = (const float*)d_in[3];
    const float* b2 = (const float*)d_in[4];
    float* out = (float*)d_out;
    _Float16* w2t = (_Float16*)d_ws;             // 5*64*256*2 = 160 KB scratch

    int B = in_sizes[0] / 2;                     // 1,048,576

    w2_transpose_kernel<<<(NLAY * 64 * HID + 255) / 256, 256, 0, stream>>>(W2, w2t);
    flow_kernel<<<B / 128, 128, 0, stream>>>(x, W1, b1, w2t, b2, out);
}
